// MultiHeadAttention_20564303414057
// MI455X (gfx1250) — compile-verified, matrix operand order checked by reading, not run on hardware
//
#include <hip/hip_runtime.h>
#include <hip/hip_bf16.h>
#include <math.h>

// ---------------------------------------------------------------------------
// Shapes (compile-time constants from the reference)
// ---------------------------------------------------------------------------
#define B_   2
#define S_   2048
#define D_   2048
#define HQ_  32
#define HKV_ 8
#define HD_  64
#define E_   3072   // (HQ_ + 2*HKV_) * HD_

typedef __attribute__((ext_vector_type(16))) __bf16        v16bf;
typedef __attribute__((ext_vector_type(8)))  __bf16        v8bf;
typedef __attribute__((ext_vector_type(8)))  float         v8f;
typedef __attribute__((ext_vector_type(4)))  unsigned int  v4u;
typedef __attribute__((ext_vector_type(8)))  int           v8i;
typedef __attribute__((ext_vector_type(4)))  int           v4i;

// Tensor Data Mover availability (device pass only; host pass lacks it)
#if defined(__has_builtin)
#  if __has_builtin(__builtin_amdgcn_tensor_load_to_lds)
#    define HAVE_TDM 1
#  endif
#endif
#ifndef HAVE_TDM
#  define HAVE_TDM 0
#endif

// ---------------------------------------------------------------------------
// WMMA helper: D = A(16x32 bf16) * B(32x16 bf16) + C(16x16 f32)
// ---------------------------------------------------------------------------
__device__ __forceinline__ v8f wmma_bf16(v16bf a, v16bf b, v8f c) {
    return __builtin_amdgcn_wmma_f32_16x16x32_bf16(
        /*neg_a=*/false, a, /*neg_b=*/false, b,
        /*c_mod=*/(short)0, c, /*reuse_a=*/false, /*reuse_b=*/false);
}

// A/B fragment loader from an fp32 row-major matrix (convert to bf16).
// ISA 16-bit 16x32 layout: lanes 0-15 hold row (lane&15), K {0..7,16..23};
// lanes 16-31 hold K {8..15,24..31}: two contiguous 8-element chunks at
// k + (lane>>4)*8 and +16.
__device__ __forceinline__ v16bf frag_from_f32(const float* __restrict__ base,
                                               int stride, int row, int k, int lane) {
    const float* p = base + (size_t)(row + (lane & 15)) * stride + k + ((lane >> 4) << 3);
    v16bf f;
#pragma unroll
    for (int i = 0; i < 8; ++i) { f[i] = (__bf16)p[i]; f[i + 8] = (__bf16)p[i + 16]; }
    return f;
}

// Same fragment loader from a bf16 row-major matrix (16B vector loads).
// Works for global and (after inlining/addrspace inference) LDS sources.
__device__ __forceinline__ v16bf frag_from_bf16(const __bf16* base,
                                                int stride, int row, int k, int lane) {
    const __bf16* p = base + (size_t)(row + (lane & 15)) * stride + k + ((lane >> 4) << 3);
    v8bf c0 = *(const v8bf*)p;
    v8bf c1 = *(const v8bf*)(p + 16);
    v16bf f;
#pragma unroll
    for (int i = 0; i < 8; ++i) { f[i] = c0[i]; f[i + 8] = c1[i]; }
    return f;
}

// ---------------------------------------------------------------------------
// TDM: issue a 2-D tile load Global -> LDS (bf16 elements, data_size=2B).
//   lds_off    : byte offset of tile start in this workgroup's LDS
//   gaddr      : global address of tile start
//   tensor_d0/1: tensor extents (elements) for OOB checking
//   tile_d0/1  : tile extents (elements); d0 is the contiguous dimension
//   stride0    : elements between consecutive d1 lines
// D# bit layout per CDNA5 ISA 8.3/8.4 (group0 128b, group1 256b).
// ---------------------------------------------------------------------------
#if HAVE_TDM
__device__ __forceinline__ void tdm_load_2d_bf16(
    unsigned lds_off, const void* gaddr,
    unsigned tensor_d0, unsigned tensor_d1,
    unsigned tile_d0, unsigned tile_d1,
    unsigned long long stride0)
{
    unsigned long long ga = (unsigned long long)(uintptr_t)gaddr;
    v4u g0;
    g0[0] = 1u;                                            // count=1 (valid), user D#
    g0[1] = lds_off;                                       // lds_addr [63:32]
    g0[2] = (unsigned)(ga & 0xffffffffu);                  // global_addr [95:64]
    g0[3] = (unsigned)((ga >> 32) & 0x01ffffffu)           // global_addr [120:96]
          | (2u << 30);                                    // type=2 ("image") [127:126]
    v8i g1;
    g1[0] = (int)(1u << 16);                               // wg_mask=0, data_size=1 (2B)
    g1[1] = (int)((tensor_d0 & 0xffffu) << 16);            // tensor_dim0[15:0] @ bits 63:48
    g1[2] = (int)((tensor_d0 >> 16) |                      // tensor_dim0[31:16]
                  ((tensor_d1 & 0xffffu) << 16));          // tensor_dim1[15:0]
    g1[3] = (int)((tensor_d1 >> 16) |                      // tensor_dim1[31:16]
                  (tile_d0 << 16));                        // tile_dim0 @ bits 127:112
    g1[4] = (int)(tile_d1 & 0xffffu);                      // tile_dim1; tile_dim2=0
    g1[5] = (int)(unsigned)(stride0 & 0xffffffffu);        // tensor_dim0_stride[31:0]
    g1[6] = (int)(unsigned)((stride0 >> 32) & 0xffffu);    // stride[47:32]; dim1_stride=0
    g1[7] = 0;
    v4i z4 = {0, 0, 0, 0};
#if __clang_major__ >= 23
    v8i z8 = {0, 0, 0, 0, 0, 0, 0, 0};
    __builtin_amdgcn_tensor_load_to_lds(g0, g1, z4, z4, z8, 0);
#else
    __builtin_amdgcn_tensor_load_to_lds(g0, g1, z4, z4, 0);
#endif
}
#endif

// ---------------------------------------------------------------------------
// Kernel 1 & 4: C(MxN,f32) = A(MxK,f32) * Bw(NxK,f32)^T using bf16 WMMA.
// Block = 8 waves as 2(M) x 4(N); wave tile = 32x64 (2x4 WMMA tiles);
// 8 WMMA per K-step from 2 A-fragments + 4 B-fragments, plus prefetch of
// the next-next K-slab (global_prefetch_b8).
// ---------------------------------------------------------------------------
__global__ __launch_bounds__(256) void gemm_f32_bf16wmma(
    const float* __restrict__ A, const float* __restrict__ Bw,
    float* __restrict__ C, int M, int N, int K)
{
    int lane = threadIdx.x & 31;
    int wave = threadIdx.x >> 5;
    int wm = wave >> 2;             // 0..1
    int wn = wave & 3;              // 0..3
    int m0 = blockIdx.x * 64  + wm * 32;
    int n0 = blockIdx.y * 256 + wn * 64;

    v8f acc[2][4] = {};
    for (int k = 0; k < K; k += 32) {
        if (k + 64 < K) {           // prefetch next-next K slab into caches
            __builtin_prefetch(A  + (size_t)(m0 + (lane & 15)) * K + k + 64, 0, 1);
            __builtin_prefetch(Bw + (size_t)(n0 + (lane & 15)) * K + k + 64, 0, 1);
        }
        v16bf a0 = frag_from_f32(A, K, m0,      k, lane);
        v16bf a1 = frag_from_f32(A, K, m0 + 16, k, lane);
#pragma unroll
        for (int t = 0; t < 4; ++t) {
            v16bf bt = frag_from_f32(Bw, K, n0 + t * 16, k, lane);
            acc[0][t] = wmma_bf16(a0, bt, acc[0][t]);
            acc[1][t] = wmma_bf16(a1, bt, acc[1][t]);
        }
    }
    // C layout: VGPR v -> row +v (lanes 0-15) / +v+8 (lanes 16-31), col lane&15
    int radd = (lane >> 4) << 3;
#pragma unroll
    for (int i = 0; i < 2; ++i)
#pragma unroll
        for (int t = 0; t < 4; ++t) {
            int cidx = n0 + t * 16 + (lane & 15);
#pragma unroll
            for (int v = 0; v < 8; ++v)
                C[(size_t)(m0 + i * 16 + v + radd) * N + cidx] = acc[i][t][v];
        }
    (void)M;
}

// ---------------------------------------------------------------------------
// Kernel 2: per-(token, head) RMSNorm + RoPE, pack to bf16.
// One wave per (token, head); lane holds dims d=lane and d=lane+32 (the
// natural RoPE rotation pair).  V heads: no norm/rope, stored TRANSPOSED
// ([b, kv, d, s]) so attention PV tiles are contiguous along s.
// ---------------------------------------------------------------------------
__global__ __launch_bounds__(256) void rmsnorm_rope_pack(
    const float* __restrict__ qkv, const float* __restrict__ qg,
    const float* __restrict__ kg,
    unsigned short* __restrict__ qb_, unsigned short* __restrict__ kb_,
    unsigned short* __restrict__ vt_)
{
    __bf16* qb = (__bf16*)qb_;
    __bf16* kb = (__bf16*)kb_;
    __bf16* vt = (__bf16*)vt_;

    int lane  = threadIdx.x & 31;
    int w     = (blockIdx.x << 3) | (threadIdx.x >> 5);   // global wave id
    int h     = w % (HQ_ + 2 * HKV_);
    int token = w / (HQ_ + 2 * HKV_);
    int s = token & (S_ - 1);
    int b = token >> 11;   // S_ == 2048

    const float* row = qkv + (size_t)token * E_ + h * HD_;
    float x1 = row[lane];
    float x2 = row[lane + 32];
    float o1 = x1, o2 = x2;

    if (h < HQ_ + HKV_) {          // q and k heads: rms-norm + rope
        float ss = x1 * x1 + x2 * x2;
#pragma unroll
        for (int msk = 1; msk < 32; msk <<= 1) ss += __shfl_xor(ss, msk, 32);
        float inv = 8.0f / fmaxf(sqrtf(ss), 1e-6f);      // sqrt(HEAD_DIM)=8
        const float* g = (h < HQ_) ? qg : kg;
        float y1 = x1 * g[lane]      * inv;
        float y2 = x2 * g[lane + 32] * inv;
        float fr = __expf(-(float)lane * 0.28782313662425575f);  // ln(1e4)/32
        float th = (float)s * fr;
        float sn, cs;
        __sincosf(th, &sn, &cs);
        o1 = y1 * cs - y2 * sn;
        o2 = y1 * sn + y2 * cs;
    }

    if (h < HQ_) {
        __bf16* d = qb + (((size_t)b * HQ_ + h) * S_ + s) * HD_;
        d[lane] = (__bf16)o1;  d[lane + 32] = (__bf16)o2;
    } else if (h < HQ_ + HKV_) {
        __bf16* d = kb + (((size_t)b * HKV_ + (h - HQ_)) * S_ + s) * HD_;
        d[lane] = (__bf16)o1;  d[lane + 32] = (__bf16)o2;
    } else {
        __bf16* d = vt + (((size_t)b * HKV_ + (h - HQ_ - HKV_)) * HD_) * (size_t)S_ + s;
        d[(size_t)lane * S_]        = (__bf16)o1;
        d[(size_t)(lane + 32) * S_] = (__bf16)o2;
    }
}

// ---------------------------------------------------------------------------
// Kernel 3: causal flash attention with DOUBLE-BUFFERED TDM staging.
// Block = (b, h, 128-query block), 8 waves x 16 query rows.  Wave 0 streams
// 32-key K (32x64 bf16) and transposed-V (64x32 bf16) chunks into ping-pong
// LDS buffers with the Tensor Data Mover; `s_wait_tensorcnt 2` (TENSORcnt
// is in-order) means the CURRENT chunk has landed while the NEXT is still
// in flight, so DMA overlaps the QK^T / softmax / PV compute.
// LDS map (bytes): [0,8192) buf0 {K,V} ; [8192,16384) buf1 {K,V} ;
//                  [16384,24576) per-wave 16x32 P tiles.
// ---------------------------------------------------------------------------
__global__ __launch_bounds__(256) void attn_fwd(
    const unsigned short* __restrict__ qb_, const unsigned short* __restrict__ kb_,
    const unsigned short* __restrict__ vt_, float* __restrict__ O)
{
    const __bf16* qb = (const __bf16*)qb_;
    const __bf16* kb = (const __bf16*)kb_;
    const __bf16* vt = (const __bf16*)vt_;

    __shared__ __bf16 smem[12288];        // 24 KB
    __bf16* sp = smem + 8192;             // P tiles: 8 x 512 elems

    int lane = threadIdx.x & 31;
    int wave = threadIdx.x >> 5;
    int qblk = blockIdx.x & 15;           // S_/128 = 16
    int h    = (blockIdx.x >> 4) & 31;
    int b    = blockIdx.x >> 9;
    int q0   = qblk * 128 + wave * 16;

    const __bf16* Qh = qb + (((size_t)b * HQ_  + h)        * S_) * HD_;
    const __bf16* Kh = kb + (((size_t)b * HKV_ + (h >> 2)) * S_) * HD_;
    const __bf16* Vt = vt + (((size_t)b * HKV_ + (h >> 2)) * HD_) * (size_t)S_;

    // Q fragments for the two K-steps of the 64-dim head
    v16bf qf0 = frag_from_bf16(Qh, HD_, q0, 0,  lane);
    v16bf qf1 = frag_from_bf16(Qh, HD_, q0, 32, lane);

    float m_r[8], l_r[8], alpha[8], p0[8], p1[8];
    v8f acc[4] = {};
#pragma unroll
    for (int v = 0; v < 8; ++v) { m_r[v] = -3.0e38f; l_r[v] = 0.0f; }

    const int col   = lane & 15;
    const int half8 = (lane >> 4) << 3;
    const float sc  = 0.125f;             // 1/sqrt(64)
    const int kend  = q0 + 16;            // this wave's causal bound
    const int nchunks = (qblk * 128 + 128) / 32;

#if HAVE_TDM
    // issue K+V tiles of chunk cc into ping-pong buffer cc&1
    auto issue_chunk = [&](int cc) {
        unsigned boff = (unsigned)(cc & 1) * 8192u;        // byte offset
        int kcc = cc * 32;
        tdm_load_2d_bf16(boff,         Kh + (size_t)kcc * HD_, HD_, S_, HD_, 32, HD_);
        tdm_load_2d_bf16(boff + 4096u, Vt + kcc,              S_,  HD_, 32, HD_, S_);
    };
    if (wave == 0) issue_chunk(0);        // prologue: start chunk 0
#endif

    for (int c = 0; c < nchunks; ++c) {
        int kc  = c * 32;
        int buf = c & 1;
        __syncthreads();                  // buffer buf^1 fully consumed by all waves
#if HAVE_TDM
        if (wave == 0) {
            if (c + 1 < nchunks) {
                issue_chunk(c + 1);                       // stream next chunk
                __builtin_amdgcn_s_wait_tensorcnt(2);     // current chunk landed
            } else {
                __builtin_amdgcn_s_wait_tensorcnt(0);
            }
        }
#else
        {   // cooperative fallback: 256 threads x 8 elems each per tile
            int t = threadIdx.x;
            __bf16* dk = smem + buf * 4096;
            __bf16* dv = smem + buf * 4096 + 2048;
            *(v8bf*)&dk[t * 8] =
                *(const v8bf*)(Kh + (size_t)(kc + (t >> 3)) * HD_ + (t & 7) * 8);
            *(v8bf*)&dv[t * 8] =
                *(const v8bf*)(Vt + (size_t)(t >> 2) * S_ + kc + (t & 3) * 8);
        }
#endif
        __syncthreads();                  // chunk c visible to everyone

        if (kc < kend) {
            const __bf16* sk = smem + buf * 4096;         // K chunk, 32x64
            const __bf16* sv = smem + buf * 4096 + 2048;  // V chunk, 64x32

            // ---- S = Q Kt : two 16x16 score tiles from the LDS K chunk ----
            v8f s0 = {}, s1 = {};
            s0 = wmma_bf16(qf0, frag_from_bf16(sk, HD_, 0,  0,  lane), s0);
            s0 = wmma_bf16(qf1, frag_from_bf16(sk, HD_, 0,  32, lane), s0);
            s1 = wmma_bf16(qf0, frag_from_bf16(sk, HD_, 16, 0,  lane), s1);
            s1 = wmma_bf16(qf1, frag_from_bf16(sk, HD_, 16, 32, lane), s1);

            // ---- online softmax (row stats per lane-half, width-16 xor) ----
#pragma unroll
            for (int v = 0; v < 8; ++v) {
                int row = q0 + v + half8;
                float a0 = (kc + col      > row) ? -3.0e38f : s0[v] * sc;
                float a1 = (kc + 16 + col > row) ? -3.0e38f : s1[v] * sc;
                float mx = fmaxf(a0, a1);
#pragma unroll
                for (int msk = 1; msk < 16; msk <<= 1)
                    mx = fmaxf(mx, __shfl_xor(mx, msk, 32));
                mx = fmaxf(mx, m_r[v]);
                alpha[v] = __expf(m_r[v] - mx);
                p0[v] = __expf(a0 - mx);
                p1[v] = __expf(a1 - mx);
                float rs = p0[v] + p1[v];
#pragma unroll
                for (int msk = 1; msk < 16; msk <<= 1)
                    rs += __shfl_xor(rs, msk, 32);
                l_r[v] = l_r[v] * alpha[v] + rs;
                m_r[v] = mx;
            }
#pragma unroll
            for (int t = 0; t < 4; ++t)
#pragma unroll
                for (int v = 0; v < 8; ++v)
                    acc[t][v] *= alpha[v];

            // ---- relayout P (C layout) -> A fragment via per-wave LDS ----
            __bf16* pt = sp + wave * 512;
#pragma unroll
            for (int v = 0; v < 8; ++v) {
                int r = v + half8;
                pt[r * 32 + col]      = (__bf16)p0[v];
                pt[r * 32 + 16 + col] = (__bf16)p1[v];
            }
            v16bf pa;
            {
                int rbase = (lane & 15) * 32 + ((lane >> 4) << 3);
#pragma unroll
                for (int i = 0; i < 8; ++i) {
                    pa[i]     = pt[rbase + i];
                    pa[i + 8] = pt[rbase + 16 + i];
                }
            }

            // ---- O += P V from the LDS V chunk (row stride 32) ----
#pragma unroll
            for (int t = 0; t < 4; ++t)
                acc[t] = wmma_bf16(pa, frag_from_bf16(sv, 32, t * 16, 0, lane), acc[t]);
        }
    }

    // ---- epilogue: normalize by l, write [b, s, h*64+d] fp32 ----
#pragma unroll
    for (int v = 0; v < 8; ++v) {
        int r = q0 + v + half8;
        float invl = 1.0f / l_r[v];
        size_t base = ((size_t)b * S_ + r) * (size_t)(HQ_ * HD_) + (size_t)h * HD_;
#pragma unroll
        for (int t = 0; t < 4; ++t)
            O[base + t * 16 + col] = acc[t][v] * invl;
    }
}

// ---------------------------------------------------------------------------
// Launcher
// ---------------------------------------------------------------------------
extern "C" void kernel_launch(void* const* d_in, const int* in_sizes, int n_in,
                              void* d_out, int out_size, void* d_ws, size_t ws_size,
                              hipStream_t stream) {
    (void)in_sizes; (void)n_in; (void)out_size; (void)ws_size;

    const float* x     = (const float*)d_in[0];
    const float* w_qkv = (const float*)d_in[1];
    const float* w_out = (const float*)d_in[2];
    const float* qg    = (const float*)d_in[3];
    const float* kg    = (const float*)d_in[4];
    float* y = (float*)d_out;

    char* ws = (char*)d_ws;
    size_t off = 0;
    float* qkv = (float*)(ws + off);                  off += (size_t)B_ * S_ * E_ * 4;          // 48 MB
    unsigned short* qb = (unsigned short*)(ws + off); off += (size_t)B_ * HQ_  * S_ * HD_ * 2;  // 16 MB
    unsigned short* kb = (unsigned short*)(ws + off); off += (size_t)B_ * HKV_ * S_ * HD_ * 2;  //  4 MB
    unsigned short* vt = (unsigned short*)(ws + off); off += (size_t)B_ * HKV_ * S_ * HD_ * 2;  //  4 MB
    float* attn = (float*)(ws + off);                 off += (size_t)B_ * S_ * HQ_ * HD_ * 4;   // 32 MB

    dim3 blk(256);

    // 1) QKV projection: qkv(4096x3072) = x(4096x2048) @ w_qkv(3072x2048)^T
    gemm_f32_bf16wmma<<<dim3(B_ * S_ / 64, E_ / 256), blk, 0, stream>>>(
        x, w_qkv, qkv, B_ * S_, E_, D_);

    // 2) per-head RMSNorm + RoPE + bf16 pack (V transposed)
    int waves = B_ * S_ * (HQ_ + 2 * HKV_);
    rmsnorm_rope_pack<<<dim3(waves / 8), blk, 0, stream>>>(qkv, qg, kg, qb, kb, vt);

    // 3) causal flash attention (GQA 4:1), double-buffered TDM K/V streaming
    attn_fwd<<<dim3(B_ * HQ_ * (S_ / 128)), blk, 0, stream>>>(qb, kb, vt, attn);

    // 4) output projection: y(4096x2048) = attn(4096x2048) @ w_out(2048x2048)^T
    gemm_f32_bf16wmma<<<dim3(B_ * S_ / 64, D_ / 256), blk, 0, stream>>>(
        attn, w_out, y, B_ * S_, D_, D_);
}
